// SwinAttention_81724637708591
// MI455X (gfx1250) — matrix operand order repaired and checked by reading, not compiled, not run on hardware
//
#include <hip/hip_runtime.h>
#include <math.h>

// ---------------------------------------------------------------------------
// Swin block (B=8, C=192, H=W=128, WS=8, shift=4, heads=4, hd=48, hidden=768)
// One workgroup (16 wave32) per (batch, window): everything lives in LDS,
// GEMMs use v_wmma_f32_16x16x32_f16 (f16 in, f32 accumulate). Weights are
// converted once per launch to f16 transposed [N][K] layout (L2-resident).
// ---------------------------------------------------------------------------

typedef __attribute__((ext_vector_type(16))) _Float16 v16h;
typedef __attribute__((ext_vector_type(8)))  _Float16 v8h;
typedef __attribute__((ext_vector_type(8)))  float    v8f;

namespace {
constexpr int kB     = 8;
constexpr int kC     = 192;
constexpr int kH     = 128;
constexpr int kW     = 128;
constexpr int kWS    = 8;
constexpr int kShift = 4;
constexpr int kHeads = 4;
constexpr int kHD    = 48;
constexpr int kL     = 64;    // tokens per window
constexpr int kN     = 256;   // windows
constexpr int kHid   = 768;
constexpr int kNpos  = 225;

constexpr int kThreads = 512; // 16 wave32 -> 4 waves per SIMD of the WGP
constexpr int kWaves   = kThreads / 32;

// f16 weight offsets (in halves) inside d_ws
constexpr int kWq_off = 0;                       // [576][192]
constexpr int kWp_off = kWq_off + 576 * 192;     // [192][192]
constexpr int kW1_off = kWp_off + 192 * 192;     // [768][192]
constexpr int kW2_off = kW1_off + 768 * 192;     // [192][768]

// LDS layout (bytes)
constexpr int kXW  = 0;                // float [64][192]  (x window, later x1)     48KB
constexpr int kXN  = 49152;            // f16   [64][192]  (LN output, A operand)   24KB
constexpr int kR   = 73728;            // reusable region:
constexpr int kQ   = kR;               // f16 [4][64][64]  Q (hd padded to 64)      32KB
constexpr int kK   = kR + 32768;       // f16 [4][64][64]  K (hd padded to 64)      32KB
constexpr int kVT  = kR + 65536;       // f16 [4][48][64]  V transposed             24KB
constexpr int kSC  = kR + 90112;       // f32 [4][64][64]  scores                   64KB
constexpr int kAT  = kR;               // f16 [4][64][64]  attn  (overlays Q)       32KB
constexpr int kAO  = kR + 32768;       // f16 [64][192]    attn out (overlays K)    24KB
constexpr int kHB  = kR;               // f16 [64][768]    mlp hidden (overlays)    96KB
constexpr int kX2  = kR + 98304;       // f32 [64][192]    final (overlays Sc tail) 48KB
constexpr int kSmemBytes = 73728 + 155648;   // 224 KB  (< 320 KB/WGP)
} // namespace

// --- WMMA fragment loaders (ISA 7.12.2 layouts, wave32) ---------------------
__device__ __forceinline__ v16h load_frag_a(const _Float16* A, int lda, int m0, int k0) {
    const int lane = threadIdx.x & 31;
    const int m = lane & 15, g = lane >> 4;
    const _Float16* p = A + (m0 + m) * lda + k0 + 8 * g;
    v8h lo = *(const v8h*)(p);        // K = 8g + 0..7
    v8h hi = *(const v8h*)(p + 16);   // K = 16 + 8g + 0..7
    v16h r;
#pragma unroll
    for (int i = 0; i < 8; ++i) { r[i] = lo[i]; r[i + 8] = hi[i]; }
    return r;
}

// B stored transposed: memory [n][k] with k contiguous
__device__ __forceinline__ v16h load_frag_b(const _Float16* Bt, int ldb, int n0, int k0) {
    const int lane = threadIdx.x & 31;
    const int n = lane & 15, g = lane >> 4;
    const _Float16* p = Bt + (n0 + n) * ldb + k0 + 8 * g;
    v8h lo = *(const v8h*)(p);        // K = 8g + 0..7
    v8h hi = *(const v8h*)(p + 16);   // K = 16 + 8g + 0..7
    v16h r;
#pragma unroll
    for (int i = 0; i < 8; ++i) { r[i] = lo[i]; r[i + 8] = hi[i]; }
    return r;
}

// PF=true: prefetch next K-panel of (global) B operand -> global_prefetch_b8
template <bool PF>
__device__ __forceinline__ v8f gemm_tile(const _Float16* A, int lda,
                                         const _Float16* Bt, int ldb,
                                         int m0, int n0, int K) {
    const int lane = threadIdx.x & 31;
    const int ln = lane & 15, lg = lane >> 4;
    v8f acc = {};
    for (int k0 = 0; k0 < K; k0 += 32) {
        if (PF && (k0 + 32 < K))
            __builtin_prefetch((const void*)(Bt + (n0 + ln) * ldb + k0 + 32 + 16 * lg), 0, 0);
        v16h a = load_frag_a(A, lda, m0, k0);
        v16h b = load_frag_b(Bt, ldb, n0, k0);
        acc = __builtin_amdgcn_wmma_f32_16x16x32_f16(false, a, false, b,
                                                     (short)0, acc, false, false);
    }
    return acc;
}

// --- LayerNorm: 8 threads per token, shfl_xor reductions --------------------
__device__ __forceinline__ void layernorm_to_f16(const float* __restrict__ src,
                                                 _Float16* __restrict__ dst,
                                                 const float* __restrict__ gam,
                                                 const float* __restrict__ bet) {
    const int tid = threadIdx.x;
    const int tok = tid >> 3;      // 0..63
    const int sl  = tid & 7;
    const float* row = src + tok * kC;
    float s = 0.f;
    for (int c = sl; c < kC; c += 8) s += row[c];
#pragma unroll
    for (int off = 4; off; off >>= 1) s += __shfl_xor(s, off, 8);
    float mu = s * (1.0f / kC);
    float vs = 0.f;
    for (int c = sl; c < kC; c += 8) { float d = row[c] - mu; vs += d * d; }
#pragma unroll
    for (int off = 4; off; off >>= 1) vs += __shfl_xor(vs, off, 8);
    float rs = rsqrtf(vs * (1.0f / kC) + 1e-5f);
    for (int c = sl; c < kC; c += 8)
        dst[tok * kC + c] = (_Float16)((row[c] - mu) * rs * gam[c] + bet[c]);
}

// --- weight prep: f32 [K][N] row-major -> f16 [N][K] ------------------------
__global__ void cvt_transpose_f16(const float* __restrict__ src,
                                  _Float16* __restrict__ dst, int K, int N) {
    int idx = blockIdx.x * blockDim.x + threadIdx.x;
    if (idx >= K * N) return;
    int k = idx / N, n = idx % N;
    dst[n * K + k] = (_Float16)src[idx];
}

// --- fused Swin block: one workgroup per (batch, window) --------------------
__global__ void __launch_bounds__(kThreads)
swin_block_kernel(const float* __restrict__ x,
                  const float* __restrict__ rel_bias,
                  const float* __restrict__ qkv_b,
                  const float* __restrict__ proj_b,
                  const float* __restrict__ n1g, const float* __restrict__ n1b,
                  const float* __restrict__ n2g, const float* __restrict__ n2b,
                  const float* __restrict__ mlp_b1, const float* __restrict__ mlp_b2,
                  const _Float16* __restrict__ wts,
                  float* __restrict__ out) {
    extern __shared__ __align__(16) unsigned char smem[];
    float*    xw = (float*)(smem + kXW);
    _Float16* xn = (_Float16*)(smem + kXN);
    _Float16* Qb = (_Float16*)(smem + kQ);
    _Float16* Kb = (_Float16*)(smem + kK);
    _Float16* Vt = (_Float16*)(smem + kVT);
    float*    Sc = (float*)(smem + kSC);
    _Float16* At = (_Float16*)(smem + kAT);
    _Float16* Ao = (_Float16*)(smem + kAO);
    _Float16* Hb = (_Float16*)(smem + kHB);
    float*    X2 = (float*)(smem + kX2);

    const int bid  = blockIdx.x;
    const int b    = bid >> 8;         // batch
    const int nw   = bid & 255;        // window
    const int wh   = nw >> 4, ww = nw & 15;
    const int tid  = threadIdx.x;
    const int wave = tid >> 5;
    const int lane = tid & 31;
    const int lg   = lane >> 4, ln = lane & 15;

    const _Float16* WTq = wts + kWq_off;
    const _Float16* WTp = wts + kWp_off;
    const _Float16* WT1 = wts + kW1_off;
    const _Float16* WT2 = wts + kW2_off;

    // ---- Phase 1: gather window (cyclic shift -4,-4) + zero Q/K pad -------
    for (int e = tid; e < kL * kC; e += kThreads) {
        int tok = e / kC, ch = e % kC;
        int r = tok >> 3, c = tok & 7;
        int gh = (wh * 8 + r + kShift) & (kH - 1);
        int gw = (ww * 8 + c + kShift) & (kW - 1);
        xw[tok * kC + ch] = x[((b * kC + ch) * kH + gh) * kW + gw];
    }
    for (int e = tid; e < kHeads * kL * 16; e += kThreads) {   // pad hd 48..63 = 0
        int head = e / (kL * 16);
        int tok  = (e >> 4) % kL;
        int pc   = 48 + (e & 15);
        Qb[(head * kL + tok) * 64 + pc] = (_Float16)0.f;
        Kb[(head * kL + tok) * 64 + pc] = (_Float16)0.f;
    }
    __syncthreads();

    // ---- LayerNorm1 -> xn (f16) --------------------------------------------
    layernorm_to_f16(xw, xn, n1g, n1b);
    __syncthreads();

    // ---- Phase 2: QKV GEMM (64x576 = 144 tiles) ----------------------------
    for (int t = wave; t < 144; t += kWaves) {
        const int ts = __builtin_amdgcn_readfirstlane(t);   // wave-uniform -> SGPR
        int m0 = (ts & 3) * 16;
        int n0 = (ts >> 2) * 16;
        v8f acc = gemm_tile<true>(xn, kC, WTq, kC, m0, n0, kC);
        int which = n0 / 192;               // 0=Q 1=K 2=V   (scalar)
        int head  = (n0 % 192) / kHD;
        int hd0   = n0 % kHD;
#pragma unroll
        for (int v = 0; v < 8; ++v) {
            int tok = m0 + v + 8 * lg;
            int col = n0 + ln;
            _Float16 hv = (_Float16)(acc[v] + qkv_b[col]);
            int hd = hd0 + ln;
            if (which == 0)      Qb[(head * kL + tok) * 64 + hd] = hv;
            else if (which == 1) Kb[(head * kL + tok) * 64 + hd] = hv;
            else                 Vt[(head * kHD + hd) * kL + tok] = hv;
        }
    }
    __syncthreads();

    // ---- Phase 3: scores = Q K^T * scale + bias/mask -----------------------
    const float scale = 0.14433756729740643f;   // 1/sqrt(48)
    for (int t = wave; t < 64; t += kWaves) {
        const int ts = __builtin_amdgcn_readfirstlane(t);
        int head = ts >> 4;
        int sub  = ts & 15;
        int m0 = (sub >> 2) * 16, n0 = (sub & 3) * 16;
        v8f acc = gemm_tile<false>(Qb + head * kL * 64, 64,
                                   Kb + head * kL * 64, 64, m0, n0, 64);
#pragma unroll
        for (int v = 0; v < 8; ++v) {
            int l = m0 + v + 8 * lg;
            int m = n0 + ln;
            int r1 = l >> 3, c1 = l & 7, r2 = m >> 3, c2 = m & 7;
            int rel = (r1 - r2 + 7) * 15 + (c1 - c2 + 7);
            float bv = rel_bias[head * kNpos + rel];
            int gr1 = wh * 8 + r1, gc1 = ww * 8 + c1;
            int gr2 = wh * 8 + r2, gc2 = ww * 8 + c2;
            int la1 = (gr1 < kH - kWS ? 0 : (gr1 < kH - kShift ? 1 : 2)) * 3 +
                      (gc1 < kW - kWS ? 0 : (gc1 < kW - kShift ? 1 : 2));
            int la2 = (gr2 < kH - kWS ? 0 : (gr2 < kH - kShift ? 1 : 2)) * 3 +
                      (gc2 < kW - kWS ? 0 : (gc2 < kW - kShift ? 1 : 2));
            float bias = (la1 != la2) ? -100.0f : bv;
            Sc[(head * kL + l) * kL + m] = acc[v] * scale + bias;
        }
    }
    __syncthreads();

    // ---- softmax: 256 rows, one per thread ---------------------------------
    if (tid < kHeads * kL) {
        int head = tid >> 6;
        int l    = tid & 63;
        float* row = Sc + (head * kL + l) * kL;
        float mx = row[0];
        for (int m = 1; m < kL; ++m) mx = fmaxf(mx, row[m]);
        float sum = 0.f;
        for (int m = 0; m < kL; ++m) { float e = __expf(row[m] - mx); row[m] = e; sum += e; }
        float inv = 1.0f / sum;
        for (int m = 0; m < kL; ++m)
            At[(head * kL + l) * kL + m] = (_Float16)(row[m] * inv);
    }
    __syncthreads();

    // ---- Phase 4: out = attn @ V  (per head 64x48) -------------------------
    for (int t = wave; t < 48; t += kWaves) {
        const int ts = __builtin_amdgcn_readfirstlane(t);
        int head = ts / 12;
        int sub  = ts % 12;
        int m0 = (sub / 3) * 16, n0 = (sub % 3) * 16;
        v8f acc = gemm_tile<false>(At + head * kL * 64, 64,
                                   Vt + head * kHD * kL, kL, m0, n0, 64);
#pragma unroll
        for (int v = 0; v < 8; ++v) {
            int tok = m0 + v + 8 * lg;
            int ch  = head * kHD + n0 + ln;
            Ao[tok * kC + ch] = (_Float16)acc[v];
        }
    }
    __syncthreads();

    // ---- Phase 5: proj GEMM + residual (x1 in-place in xw) -----------------
    for (int t = wave; t < 48; t += kWaves) {
        const int ts = __builtin_amdgcn_readfirstlane(t);
        int m0 = (ts & 3) * 16, n0 = (ts >> 2) * 16;
        v8f acc = gemm_tile<true>(Ao, kC, WTp, kC, m0, n0, kC);
#pragma unroll
        for (int v = 0; v < 8; ++v) {
            int tok = m0 + v + 8 * lg;
            int col = n0 + ln;
            xw[tok * kC + col] += acc[v] + proj_b[col];
        }
    }
    __syncthreads();

    // ---- LayerNorm2 -> xn --------------------------------------------------
    layernorm_to_f16(xw, xn, n2g, n2b);
    __syncthreads();

    // ---- Phase 7: MLP1 + exact GELU (64x768 = 192 tiles) -------------------
    for (int t = wave; t < 192; t += kWaves) {
        const int ts = __builtin_amdgcn_readfirstlane(t);
        int m0 = (ts & 3) * 16, n0 = (ts >> 2) * 16;
        v8f acc = gemm_tile<true>(xn, kC, WT1, kC, m0, n0, kC);
#pragma unroll
        for (int v = 0; v < 8; ++v) {
            int tok = m0 + v + 8 * lg;
            int col = n0 + ln;
            float val = acc[v] + mlp_b1[col];
            float ge  = 0.5f * val * (1.0f + erff(val * 0.70710678118654752f));
            Hb[tok * kHid + col] = (_Float16)ge;
        }
    }
    __syncthreads();

    // ---- Phase 8: MLP2 + residual -> X2 ------------------------------------
    for (int t = wave; t < 48; t += kWaves) {
        const int ts = __builtin_amdgcn_readfirstlane(t);
        int m0 = (ts & 3) * 16, n0 = (ts >> 2) * 16;
        v8f acc = gemm_tile<true>(Hb, kHid, WT2, kHid, m0, n0, kHid);
#pragma unroll
        for (int v = 0; v < 8; ++v) {
            int tok = m0 + v + 8 * lg;
            int col = n0 + ln;
            X2[tok * kC + col] = acc[v] + mlp_b2[col] + xw[tok * kC + col];
        }
    }
    __syncthreads();

    // ---- window reverse + roll(+4,+4), 32B-contiguous runs -----------------
    for (int e = tid; e < kC * kL; e += kThreads) {
        int ch  = e / kL;
        int tok = e % kL;
        int r = tok >> 3, c = tok & 7;
        int gh = (wh * 8 + r + kShift) & (kH - 1);
        int gw = (ww * 8 + c + kShift) & (kW - 1);
        out[((b * kC + ch) * kH + gh) * kW + gw] = X2[tok * kC + ch];
    }
}

// ---------------------------------------------------------------------------
extern "C" void kernel_launch(void* const* d_in, const int* in_sizes, int n_in,
                              void* d_out, int out_size, void* d_ws, size_t ws_size,
                              hipStream_t stream) {
    (void)in_sizes; (void)n_in; (void)out_size; (void)ws_size;
    const float* x        = (const float*)d_in[0];
    const float* rel_bias = (const float*)d_in[1];
    const float* qkv_w    = (const float*)d_in[2];
    const float* qkv_b    = (const float*)d_in[3];
    const float* proj_w   = (const float*)d_in[4];
    const float* proj_b   = (const float*)d_in[5];
    const float* n1g      = (const float*)d_in[6];
    const float* n1b      = (const float*)d_in[7];
    const float* n2g      = (const float*)d_in[8];
    const float* n2b      = (const float*)d_in[9];
    const float* mlp_w1   = (const float*)d_in[10];
    const float* mlp_b1   = (const float*)d_in[11];
    const float* mlp_w2   = (const float*)d_in[12];
    const float* mlp_b2   = (const float*)d_in[13];
    float* out            = (float*)d_out;
    _Float16* wts         = (_Float16*)d_ws;

    // Weight prep: f32 [K][N] -> f16 [N][K] (transposed for WMMA B-operand)
    cvt_transpose_f16<<<(192 * 576 + 255) / 256, 256, 0, stream>>>(qkv_w,  wts + kWq_off, 192, 576);
    cvt_transpose_f16<<<(192 * 192 + 255) / 256, 256, 0, stream>>>(proj_w, wts + kWp_off, 192, 192);
    cvt_transpose_f16<<<(192 * 768 + 255) / 256, 256, 0, stream>>>(mlp_w1, wts + kW1_off, 192, 768);
    cvt_transpose_f16<<<(768 * 192 + 255) / 256, 256, 0, stream>>>(mlp_w2, wts + kW2_off, 768, 192);

    // One workgroup (16 wave32) per (batch, window)
    swin_block_kernel<<<kB * kN, kThreads, kSmemBytes, stream>>>(
        x, rel_bias, qkv_b, proj_b, n1g, n1b, n2g, n2b, mlp_b1, mlp_b2, wts, out);
}
